// CausalSelfAttention_64106681860283
// MI455X (gfx1250) — matrix operand order repaired and checked by reading, not compiled, not run on hardware
//
#include <hip/hip_runtime.h>
#include <hip/hip_bf16.h>
#include <stdint.h>

// Problem constants
#define BB 4
#define TT 2048
#define CC 1024
#define NHH 16
#define HDD 64
#define MROWS (BB * TT)   // 8192
#define N_QKV (3 * CC)    // 3072

typedef __attribute__((ext_vector_type(16))) __bf16 v16bf;
typedef __attribute__((ext_vector_type(8)))  float  v8f;

union FragU {
    uint4        q[2];
    unsigned int u[8];
    v16bf        v;
};

__device__ __forceinline__ unsigned short f2bf(float f) {
    union { float f; unsigned int u; } x; x.f = f;
    unsigned int u = x.u;
    unsigned int r = (u + 0x7FFFu + ((u >> 16) & 1u)) >> 16;  // RNE
    return (unsigned short)r;
}
__device__ __forceinline__ unsigned int pack_bf2(float lo, float hi) {
    return (unsigned int)f2bf(lo) | ((unsigned int)f2bf(hi) << 16);
}

// Async global->LDS b128 copy (CDNA5). saddr form:
//   mem addr = SGPR64 base + VGPR 32-bit byte offset
//   LDS addr = VGPR 32-bit byte offset (workgroup LDS base applied by HW)
__device__ __forceinline__ void async_ld_b128(unsigned lds_off,
                                              unsigned gmem_off,
                                              unsigned long long base) {
    asm volatile("global_load_async_to_lds_b128 %0, %1, %2"
                 :
                 : "v"(lds_off), "v"(gmem_off), "s"(base)
                 : "memory");
}
__device__ __forceinline__ void wait_async_le4() {
    asm volatile("s_wait_asynccnt 0x4" ::: "memory");
}
__device__ __forceinline__ void wait_async_le0() {
    asm volatile("s_wait_asynccnt 0x0" ::: "memory");
}

// ---------------------------------------------------------------------------
// Stage 0a: plain f32 -> bf16 conversion (4 elems / thread)
// ---------------------------------------------------------------------------
__global__ void cvt_f32_bf16_kernel(const float* __restrict__ src,
                                    unsigned int* __restrict__ dst, int n) {
    int i = (blockIdx.x * blockDim.x + threadIdx.x) * 4;
    if (i + 3 < n) {
        float4 f = *(const float4*)(src + i);
        uint2 o;
        o.x = pack_bf2(f.x, f.y);
        o.y = pack_bf2(f.z, f.w);
        *(uint2*)(dst + (i >> 1)) = o;
    }
}

// ---------------------------------------------------------------------------
// Stage 0b: transpose+convert: src[K][N] f32  ->  dst[N][K] bf16
// ---------------------------------------------------------------------------
__global__ void cvtT_f32_bf16_kernel(const float* __restrict__ src,
                                     unsigned short* __restrict__ dst,
                                     int K, int N) {
    long i = (long)blockIdx.x * blockDim.x + threadIdx.x;
    long total = (long)K * N;
    if (i < total) {
        int n = (int)(i / K);
        int k = (int)(i % K);
        dst[i] = f2bf(src[(long)k * N + n]);
    }
}

// ---------------------------------------------------------------------------
// Stage 1: QKV = x @ w_attn + b_attn.
// A = x_bf16 [M][C]. B = w_attnT [3C][C] (pre-transposed).
// Block 128 threads = 4 waves (2x2), block tile 64x128, wave tile 32x64:
// 8 WMMA per 12 b128 fragment loads per K-step.
// Epilogue scatters Q (scaled 1/sqrt(HD)), K as [b,h,T,HD], V as [b,h,HD,T].
// ---------------------------------------------------------------------------
__global__ __launch_bounds__(128)
void qkv_gemm_kernel(const unsigned short* __restrict__ xb,
                     const unsigned short* __restrict__ waT,
                     const float* __restrict__ b_attn,
                     unsigned short* __restrict__ Qb,
                     unsigned short* __restrict__ Kb,
                     unsigned short* __restrict__ VT) {
    const int lane = threadIdx.x & 31;
    const int wave = threadIdx.x >> 5;
    const int hh = lane >> 4;
    const int lr = lane & 15;
    const int m0 = blockIdx.y * 64 + (wave >> 1) * 32;
    const int n0 = blockIdx.x * 128 + (wave & 1) * 64;

    v8f acc[2][4];
#pragma unroll
    for (int mi = 0; mi < 2; ++mi)
#pragma unroll
        for (int ni = 0; ni < 4; ++ni) acc[mi][ni] = (v8f)(0.0f);

    for (int kk = 0; kk < CC; kk += 32) {
        FragU a[2], b[4];
#pragma unroll
        for (int mi = 0; mi < 2; ++mi) {
            const unsigned short* p =
                xb + (size_t)(m0 + mi * 16 + lr) * CC + kk + hh * 8;
            a[mi].q[0] = *(const uint4*)(p);
            a[mi].q[1] = *(const uint4*)(p + 16);
        }
#pragma unroll
        for (int ni = 0; ni < 4; ++ni) {
            const unsigned short* p =
                waT + (size_t)(n0 + ni * 16 + lr) * CC + kk + hh * 8;
            b[ni].q[0] = *(const uint4*)(p);
            b[ni].q[1] = *(const uint4*)(p + 16);
        }
#pragma unroll
        for (int mi = 0; mi < 2; ++mi)
#pragma unroll
            for (int ni = 0; ni < 4; ++ni)
                acc[mi][ni] = __builtin_amdgcn_wmma_f32_16x16x32_bf16(
                    false, a[mi].v, false, b[ni].v, (short)0, acc[mi][ni],
                    false, false);
    }

#pragma unroll
    for (int mi = 0; mi < 2; ++mi) {
#pragma unroll
        for (int ni = 0; ni < 4; ++ni) {
            const int n = n0 + ni * 16 + lr;
            const float bias = b_attn[n];
            const int which = n >> 10;          // 0=q 1=k 2=v
            const int cm = n & (CC - 1);
            const int head = cm >> 6;
            const int hd = cm & 63;
#pragma unroll
            for (int v = 0; v < 8; ++v) {
                const int m = m0 + mi * 16 + hh * 8 + v;
                const int bidx = m >> 11;       // / TT
                const int t = m & (TT - 1);
                const float val = acc[mi][ni][v] + bias;
                if (which == 0) {
                    Qb[(((size_t)(bidx * NHH + head) * TT + t) * HDD + hd)] =
                        f2bf(val * 0.125f);
                } else if (which == 1) {
                    Kb[(((size_t)(bidx * NHH + head) * TT + t) * HDD + hd)] =
                        f2bf(val);
                } else {
                    VT[(((size_t)(bidx * NHH + head) * HDD + hd) * TT + t)] =
                        f2bf(val);
                }
            }
        }
    }
}

// ---------------------------------------------------------------------------
// Stage 2: causal flash attention with async-to-LDS K/V staging.
// Block = 128 threads = 4 waves; each wave owns 16 queries (block: 64).
// K/V tiles (32 keys) are double-buffered in LDS via
// GLOBAL_LOAD_ASYNC_TO_LDS_B128 (4 issues/thread/tile, ASYNCcnt tracked).
// Scores computed as S^T = K @ Q^T so exp'd probabilities are already in
// the A-fragment layout for P@V (zero cross-lane transpose).
// ---------------------------------------------------------------------------
#define KP 72   // padded K-tile row (bf16 elems): 32 rows x 64 data + 8 pad
#define VP 40   // padded V-tile row (bf16 elems): 64 rows x 32 data + 8 pad

__global__ __launch_bounds__(128)
void attn_kernel(const unsigned short* __restrict__ Qb,
                 const unsigned short* __restrict__ Kb,
                 const unsigned short* __restrict__ VT,
                 unsigned short* __restrict__ AO) {
    __shared__ unsigned short ksh[2][32][KP];
    __shared__ unsigned short vsh[2][64][VP];

    const int tid = threadIdx.x;
    const int lane = tid & 31;
    const int wave = tid >> 5;      // 0..3
    const int hh = lane >> 4;
    const int lr = lane & 15;
    const int b = blockIdx.z;
    const int h = blockIdx.y;
    const int q0_blk = blockIdx.x * 64;
    const int q0 = q0_blk + wave * 16;

    const unsigned short* Qp = Qb + (size_t)(b * NHH + h) * TT * HDD;
    const unsigned long long Kbase =
        (unsigned long long)(size_t)(Kb + (size_t)(b * NHH + h) * TT * HDD);
    const unsigned long long Vbase =
        (unsigned long long)(size_t)(VT + (size_t)(b * NHH + h) * HDD * TT);

    // Q^T B-fragments for the two HD chunks (K=0..31, K=32..63)
    FragU qf[2];
    {
        const unsigned short* p = Qp + (size_t)(q0 + lr) * HDD + hh * 8;
        qf[0].q[0] = *(const uint4*)(p);
        qf[0].q[1] = *(const uint4*)(p + 16);
        qf[1].q[0] = *(const uint4*)(p + 32);
        qf[1].q[1] = *(const uint4*)(p + 48);
    }

    float m_run = -INFINITY;
    float l_run = 0.0f;
    v8f o[4];
#pragma unroll
    for (int j = 0; j < 4; ++j) o[j] = (v8f)(0.0f);

    const int q_glob = q0 + lr;                   // lane's query (S^T layout)
    const int ntiles = ((q0_blk + 63) >> 5) + 1;  // block-uniform

    // issue async copies of K/V tile kt into LDS buffer bb (4 per thread)
    auto issue = [&](int kt, int bb) {
        const int k0 = kt * 32;
#pragma unroll
        for (int i = 0; i < 2; ++i) {
            // K tile: 32 rows x 128B = 256 x b128 chunks
            const int ci = tid + i * 128;
            const int kr = ci >> 3, kc = ci & 7;              // row, 16B col
            async_ld_b128((unsigned)(size_t)&ksh[bb][kr][kc * 8],
                          (unsigned)(((k0 + kr) * HDD + kc * 8) * 2), Kbase);
            // V^T tile: 64 rows x 64B = 256 x b128 chunks
            const int vr = ci >> 2, vc = ci & 3;
            async_ld_b128((unsigned)(size_t)&vsh[bb][vr][vc * 8],
                          (unsigned)((vr * TT + k0 + vc * 8) * 2), Vbase);
        }
    };

    issue(0, 0);
    for (int kt = 0; kt < ntiles; ++kt) {
        const int bb = kt & 1;
        const int k0 = kt * 32;
        if (kt + 1 < ntiles) {
            issue(kt + 1, bb ^ 1);
            wait_async_le4();   // tile kt's 4 copies retired (in-order)
        } else {
            wait_async_le0();
        }
        __syncthreads();

        if (k0 <= q0 + 15) {   // wave-uniform causal skip; EXEC stays full
            // K A-fragments from LDS: [t]=key subtile, [c]=HD chunk
            FragU ka[2][2];
#pragma unroll
            for (int t = 0; t < 2; ++t) {
                const unsigned short* p = &ksh[bb][t * 16 + lr][hh * 8];
                ka[t][0].q[0] = *(const uint4*)(p);
                ka[t][0].q[1] = *(const uint4*)(p + 16);
                ka[t][1].q[0] = *(const uint4*)(p + 32);
                ka[t][1].q[1] = *(const uint4*)(p + 48);
            }

            // S^T = K @ Q^T (two 16-key subtiles, accumulate over HD)
            v8f st[2];
#pragma unroll
            for (int t = 0; t < 2; ++t) {
                v8f z = (v8f)(0.0f);
                st[t] = __builtin_amdgcn_wmma_f32_16x16x32_bf16(
                    false, ka[t][0].v, false, qf[0].v, (short)0, z, false,
                    false);
                st[t] = __builtin_amdgcn_wmma_f32_16x16x32_bf16(
                    false, ka[t][1].v, false, qf[1].v, (short)0, st[t], false,
                    false);
            }

            // causal mask (key > query -> -inf)
            if (k0 + 31 > q0) {
#pragma unroll
                for (int t = 0; t < 2; ++t)
#pragma unroll
                    for (int v = 0; v < 8; ++v) {
                        const int key = k0 + t * 16 + hh * 8 + v;
                        st[t][v] = (key > q_glob) ? -INFINITY : st[t][v];
                    }
            }

            // online softmax (stats at lane==query, replicated across halves)
            float mx = st[0][0];
#pragma unroll
            for (int v = 1; v < 8; ++v) mx = fmaxf(mx, st[0][v]);
#pragma unroll
            for (int v = 0; v < 8; ++v) mx = fmaxf(mx, st[1][v]);
            mx = fmaxf(mx, __shfl_xor(mx, 16));
            const float m_new = fmaxf(m_run, mx);
            const float scale = __expf(m_run - m_new);

            float p0[8], p1[8];
            float rs = 0.0f;
#pragma unroll
            for (int v = 0; v < 8; ++v) { p0[v] = __expf(st[0][v] - m_new); rs += p0[v]; }
#pragma unroll
            for (int v = 0; v < 8; ++v) { p1[v] = __expf(st[1][v] - m_new); rs += p1[v]; }
            rs += __shfl_xor(rs, 16);
            l_run = l_run * scale + rs;
            m_run = m_new;

            // P as A-fragment (16 queries x 32 keys) — direct repack
            FragU pf;
#pragma unroll
            for (int v = 0; v < 4; ++v) pf.u[v] = pack_bf2(p0[2 * v], p0[2 * v + 1]);
#pragma unroll
            for (int v = 0; v < 4; ++v) pf.u[4 + v] = pack_bf2(p1[2 * v], p1[2 * v + 1]);

            // per-query rescale broadcast into O layout (row = hh*8+v)
            float scq[8];
#pragma unroll
            for (int v = 0; v < 8; ++v) scq[v] = __shfl(scale, hh * 8 + v);

            // O += P @ V over 4 HD subtiles; V^T tile gives table-order B-frags
#pragma unroll
            for (int j = 0; j < 4; ++j) {
                FragU vb;
                const unsigned short* p = &vsh[bb][j * 16 + lr][hh * 8];
                vb.q[0] = *(const uint4*)(p);
                vb.q[1] = *(const uint4*)(p + 16);
                v8f c;
#pragma unroll
                for (int v = 0; v < 8; ++v) c[v] = o[j][v] * scq[v];
                o[j] = __builtin_amdgcn_wmma_f32_16x16x32_bf16(
                    false, pf.v, false, vb.v, (short)0, c, false, false);
            }
        }
        __syncthreads();   // before buffer bb is overwritten
    }

    // finalize: divide by l per query, write [B,T,C] bf16
    const float linv = 1.0f / l_run;
    float li[8];
#pragma unroll
    for (int v = 0; v < 8; ++v) li[v] = __shfl(linv, hh * 8 + v);
#pragma unroll
    for (int j = 0; j < 4; ++j)
#pragma unroll
        for (int v = 0; v < 8; ++v) {
            const int t = q0 + hh * 8 + v;
            AO[(size_t)(b * TT + t) * CC + h * HDD + j * 16 + lr] =
                f2bf(o[j][v] * li[v]);
        }
}

// ---------------------------------------------------------------------------
// Stage 3: out = attn_out @ w_proj + b_proj  (f32 output)
// Same tiling as stage 1; B = w_projT [C][C] pre-transposed bf16.
// ---------------------------------------------------------------------------
__global__ __launch_bounds__(128)
void proj_gemm_kernel(const unsigned short* __restrict__ AO,
                      const unsigned short* __restrict__ wpT,
                      const float* __restrict__ b_proj,
                      float* __restrict__ out) {
    const int lane = threadIdx.x & 31;
    const int wave = threadIdx.x >> 5;
    const int hh = lane >> 4;
    const int lr = lane & 15;
    const int m0 = blockIdx.y * 64 + (wave >> 1) * 32;
    const int n0 = blockIdx.x * 128 + (wave & 1) * 64;

    v8f acc[2][4];
#pragma unroll
    for (int mi = 0; mi < 2; ++mi)
#pragma unroll
        for (int ni = 0; ni < 4; ++ni) acc[mi][ni] = (v8f)(0.0f);

    for (int kk = 0; kk < CC; kk += 32) {
        FragU a[2], b[4];
#pragma unroll
        for (int mi = 0; mi < 2; ++mi) {
            const unsigned short* p =
                AO + (size_t)(m0 + mi * 16 + lr) * CC + kk + hh * 8;
            a[mi].q[0] = *(const uint4*)(p);
            a[mi].q[1] = *(const uint4*)(p + 16);
        }
#pragma unroll
        for (int ni = 0; ni < 4; ++ni) {
            const unsigned short* p =
                wpT + (size_t)(n0 + ni * 16 + lr) * CC + kk + hh * 8;
            b[ni].q[0] = *(const uint4*)(p);
            b[ni].q[1] = *(const uint4*)(p + 16);
        }
#pragma unroll
        for (int mi = 0; mi < 2; ++mi)
#pragma unroll
            for (int ni = 0; ni < 4; ++ni)
                acc[mi][ni] = __builtin_amdgcn_wmma_f32_16x16x32_bf16(
                    false, a[mi].v, false, b[ni].v, (short)0, acc[mi][ni],
                    false, false);
    }

#pragma unroll
    for (int mi = 0; mi < 2; ++mi)
#pragma unroll
        for (int ni = 0; ni < 4; ++ni) {
            const int n = n0 + ni * 16 + lr;
            const float bias = b_proj[n];
#pragma unroll
            for (int v = 0; v < 8; ++v) {
                const int m = m0 + mi * 16 + hh * 8 + v;
                out[(size_t)m * CC + n] = acc[mi][ni][v] + bias;
            }
        }
}

// ---------------------------------------------------------------------------
// Host launcher
// ---------------------------------------------------------------------------
extern "C" void kernel_launch(void* const* d_in, const int* in_sizes, int n_in,
                              void* d_out, int out_size, void* d_ws,
                              size_t ws_size, hipStream_t stream) {
    const float* x      = (const float*)d_in[0];   // [B,T,C]
    const float* w_attn = (const float*)d_in[1];   // [C,3C]
    const float* b_attn = (const float*)d_in[2];   // [3C]
    const float* w_proj = (const float*)d_in[3];   // [C,C]
    const float* b_proj = (const float*)d_in[4];   // [C]
    float* out = (float*)d_out;

    char* ws = (char*)d_ws;
    const size_t MB = 1024ull * 1024ull;
    unsigned short* xb  = (unsigned short*)(ws);              // 16 MB
    unsigned short* waT = (unsigned short*)(ws + 16 * MB);    // 6 MB
    unsigned short* wpT = (unsigned short*)(ws + 22 * MB);    // 2 MB
    unsigned short* Qb  = (unsigned short*)(ws + 24 * MB);    // 16 MB
    unsigned short* Kb  = (unsigned short*)(ws + 40 * MB);    // 16 MB
    unsigned short* VT  = (unsigned short*)(ws + 56 * MB);    // 16 MB
    unsigned short* AO  = (unsigned short*)(ws + 72 * MB);    // 16 MB

    // Stage 0: precision / layout conversions
    {
        const int n = MROWS * CC;  // 8388608
        cvt_f32_bf16_kernel<<<n / (256 * 4), 256, 0, stream>>>(
            x, (unsigned int*)xb, n);
    }
    {
        const long total = (long)CC * N_QKV;  // 3145728
        cvtT_f32_bf16_kernel<<<(int)((total + 255) / 256), 256, 0, stream>>>(
            w_attn, waT, CC, N_QKV);
    }
    {
        const long total = (long)CC * CC;  // 1048576
        cvtT_f32_bf16_kernel<<<(int)((total + 255) / 256), 256, 0, stream>>>(
            w_proj, wpT, CC, CC);
    }

    // Stage 1: QKV GEMM
    qkv_gemm_kernel<<<dim3(N_QKV / 128, MROWS / 64), 128, 0, stream>>>(
        xb, waT, b_attn, Qb, Kb, VT);

    // Stage 2: causal flash attention (async LDS staging)
    attn_kernel<<<dim3(TT / 64, NHH, BB), 128, 0, stream>>>(Qb, Kb, VT, AO);

    // Stage 3: output projection
    proj_gemm_kernel<<<dim3(CC / 128, MROWS / 64), 128, 0, stream>>>(
        AO, wpT, b_proj, out);
}
